// SwinTransformerBlock3D_83562883711261
// MI455X (gfx1250) — compile-verified, matrix operand order checked by reading, not run on hardware
//
#include <hip/hip_runtime.h>
#include <hip/hip_bf16.h>
#include <math.h>

// ---------------------------------------------------------------------------
// SwinTransformerBlock3D for MI455X (gfx1250, wave32, WMMA 16x16x32 f16)
// B=2 D=32 H=56 W=56 C=96, windows (2,7,7) -> 2048 windows x 98 tokens (pad 112)
// heads=3, hd=32.  All GEMM contractions via v_wmma_f32_16x16x32_f16.
// Streamed activations get global_prefetch hints (ISA 10.5) one K-step ahead.
// ---------------------------------------------------------------------------

typedef __attribute__((ext_vector_type(16))) _Float16 v16h;
typedef __attribute__((ext_vector_type(8)))  float    v8f;

#define WMMA_F16(a, b, c) \
  __builtin_amdgcn_wmma_f32_16x16x32_f16(false, (a), false, (b), (short)0, (c), false, false)

// K-pair base for 16-bit A/B fragments (ISA 7.12.2: 16-bit A 16x32 layout).
// VGPR j holds K = {kb, kb+1}; lanes 0-15 row m (halves K 0-7,16-23),
// lanes 16-31 same rows (halves K 8-15,24-31).
__device__ __forceinline__ int frag_kbase(int j, int hf) {
  return ((j & 4) ? 16 : 0) + ((j & 3) << 1) + (hf << 3);
}

// Load a 16x32 A fragment (or a B fragment from row-major [N x K] weight
// storage, since B column n == weight row n) from a row-major tile.
__device__ __forceinline__ v16h load_frag_rm(const _Float16* __restrict__ base, int ld) {
  const int lane = threadIdx.x & 31;
  const int r = lane & 15, hf = lane >> 4;
  v16h f;
#pragma unroll
  for (int j = 0; j < 8; ++j) {
    const int kb = frag_kbase(j, hf);
    f[2 * j]     = base[r * ld + kb];
    f[2 * j + 1] = base[r * ld + kb + 1];
  }
  return f;
}

// B fragment for P@V from Vt stored as [32 x 112] (d-major), col tile ncol0,
// K offset k0, guarded at K=112 (K padded to 128 with zeros).
__device__ __forceinline__ v16h load_frag_vt(const _Float16* __restrict__ vt,
                                             int ncol0, int k0) {
  const int lane = threadIdx.x & 31;
  const int n = lane & 15, hf = lane >> 4;
  const _Float16* rowp = vt + (ncol0 + n) * 112;
  v16h f;
#pragma unroll
  for (int j = 0; j < 8; ++j) {
    const int kb = k0 + frag_kbase(j, hf);
    f[2 * j]     = (kb     < 112) ? rowp[kb]     : (_Float16)0.f;
    f[2 * j + 1] = (kb + 1 < 112) ? rowp[kb + 1] : (_Float16)0.f;
  }
  return f;
}

// Per-lane prefetch of a 16-row streamed A tile, one byte per lane covers the
// cachelines (global_prefetch_b8, ISA 10.5); rows split across the 32 lanes.
__device__ __forceinline__ void prefetch_tile_rows(const _Float16* base, int ld) {
  const int lane = threadIdx.x & 31;
  const int r = lane & 15, hx = (lane >> 4) << 4;  // 2 probes per row, 32B apart
  __builtin_prefetch(base + r * ld + hx, 0, 3);
}

// --------------------------- constants -------------------------------------
#define NB   2
#define ND   32
#define NH   56
#define NW   56
#define NC   96
#define NWIN 2048           // total windows (B * 1024)
#define NTOK 112            // padded tokens per window (98 valid)
#define MROWS (NWIN * NTOK) // 229376 padded token-rows
#define LTOK  200704        // B*D*H*W real tokens
#define QSCALE 0.17677669529663687f

// ------------------- kernel 0: weights fp32 -> f16 --------------------------
__global__ void k_cvt_w(const float* __restrict__ qw, const float* __restrict__ pw,
                        const float* __restrict__ f1, const float* __restrict__ f2,
                        _Float16* wq, _Float16* wp, _Float16* w1, _Float16* w2) {
  const int i = blockIdx.x * 256 + threadIdx.x;
  if (i < 27648) wq[i] = (_Float16)qw[i];
  if (i < 9216)  wp[i] = (_Float16)pw[i];
  if (i < 36864) { w1[i] = (_Float16)f1[i]; w2[i] = (_Float16)f2[i]; }
}

// ------ kernel 1: LN1 + roll(-shift) + window partition -> xw f16 -----------
__global__ void k_ln1_window(const float* __restrict__ x, const float* __restrict__ g,
                             const float* __restrict__ bt, _Float16* __restrict__ xw) {
  const int lane = threadIdx.x & 31;
  const int unit = blockIdx.x * 4 + (threadIdx.x >> 5);  // one wave per token
  if (unit >= MROWS) return;
  const int win = unit / NTOK, tok = unit % NTOK;
  _Float16* dst = xw + (long)unit * NC;
  if (tok >= 98) {                                       // zero-pad rows
    for (int c = lane; c < NC; c += 32) dst[c] = (_Float16)0.f;
    return;
  }
  int b = win >> 10, rem = win & 1023;
  const int wd = rem >> 6; rem &= 63;
  const int wh = rem >> 3, ww = rem & 7;
  const int td = tok / 49, r2 = tok % 49, th = r2 / 7, tw = r2 % 7;
  const int d = ((wd << 1) + td + 1) & 31;               // gather = roll(-s)
  const int h = (wh * 7 + th + 3) % NH;
  const int w = (ww * 7 + tw + 3) % NW;
  const float* src = x + ((((long)b * ND + d) * NH + h) * NW + w) * NC;
  const float v0 = src[lane], v1 = src[lane + 32], v2 = src[lane + 64];
  float s = v0 + v1 + v2, ss = v0 * v0 + v1 * v1 + v2 * v2;
#pragma unroll
  for (int off = 16; off > 0; off >>= 1) {
    s  += __shfl_xor(s, off, 32);
    ss += __shfl_xor(ss, off, 32);
  }
  const float mean = s * (1.f / 96.f);
  const float var  = ss * (1.f / 96.f) - mean * mean;
  const float rstd = rsqrtf(var + 1e-5f);
  dst[lane]      = (_Float16)((v0 - mean) * rstd * g[lane]      + bt[lane]);
  dst[lane + 32] = (_Float16)((v1 - mean) * rstd * g[lane + 32] + bt[lane + 32]);
  dst[lane + 64] = (_Float16)((v2 - mean) * rstd * g[lane + 64] + bt[lane + 64]);
}

// ------ kernel 2: QKV GEMM (M=229376, N=288, K=96) -> q*scale, k, v^T -------
__global__ void k_qkv(const _Float16* __restrict__ xw, const _Float16* __restrict__ wq,
                      const float* __restrict__ bq, _Float16* __restrict__ q,
                      _Float16* __restrict__ kk, _Float16* __restrict__ vt) {
  const int lane = threadIdx.x & 31, wv = threadIdx.x >> 5;
  const int n = lane & 15, hf = lane >> 4;
  const int m0 = (blockIdx.x * 4 + wv) * 16;
  const int n0 = blockIdx.y * 48;
  prefetch_tile_rows(xw + (long)m0 * 96, 96);
  v8f a0 = {}, a1 = {}, a2 = {};
#pragma unroll
  for (int kkk = 0; kkk < 96; kkk += 32) {
    const v16h a  = load_frag_rm(xw + (long)m0 * 96 + kkk, 96);
    const v16h b0 = load_frag_rm(wq + (n0 +  0) * 96 + kkk, 96);
    const v16h b1 = load_frag_rm(wq + (n0 + 16) * 96 + kkk, 96);
    const v16h b2 = load_frag_rm(wq + (n0 + 32) * 96 + kkk, 96);
    a0 = WMMA_F16(a, b0, a0);
    a1 = WMMA_F16(a, b1, a1);
    a2 = WMMA_F16(a, b2, a2);
  }
#pragma unroll
  for (int t = 0; t < 3; ++t) {
    const v8f acc = (t == 0) ? a0 : (t == 1) ? a1 : a2;
    const int col = n0 + t * 16 + n;
    const float bias = bq[col];
    const int e = col / 96, rem = col % 96, head = rem >> 5, dd = rem & 31;
#pragma unroll
    for (int r = 0; r < 8; ++r) {
      const int row = m0 + r + 8 * hf;
      const int win = row / NTOK, tok = row % NTOK;
      const float val = acc[r] + bias;
      const long base = ((long)(win * 3 + head) * NTOK + tok) * 32 + dd;
      if (e == 0)      q[base]  = (_Float16)(val * QSCALE);
      else if (e == 1) kk[base] = (_Float16)val;
      else vt[((long)(win * 3 + head) * 32 + dd) * NTOK + tok] = (_Float16)val;
    }
  }
}

// ------ kernel 3: fused attention per (window, head): S=QK^T, softmax, PV ---
__global__ void __launch_bounds__(224)
k_attn(const _Float16* __restrict__ q, const _Float16* __restrict__ kk,
       const _Float16* __restrict__ vt, const float* __restrict__ mask,
       _Float16* __restrict__ o) {
  __shared__ _Float16 P[7][16][128];               // 28 KB: per-wave P rows, K padded
  const int lane = threadIdx.x & 31, wv = threadIdx.x >> 5;
  const int n = lane & 15, hf = lane >> 4;
  const int unit = blockIdx.x;                     // win*3+head
  const int win = unit / 3;
  const int mw = win & 1023;
  const long qkbase = (long)unit * NTOK * 32;
  const int m0 = wv * 16;

  const v16h aq = load_frag_rm(q + qkbase + (long)m0 * 32, 32);
  v8f acc[7];
#pragma unroll
  for (int ct = 0; ct < 7; ++ct) {
    const v16h bk = load_frag_rm(kk + qkbase + (long)(ct * 16) * 32, 32);
    v8f z = {};
    acc[ct] = WMMA_F16(aq, bk, z);
  }
  // masked softmax per row (row strided over 16-lane half-wave groups)
#pragma unroll
  for (int r = 0; r < 8; ++r) {
    const int row = m0 + r + 8 * hf;
    float sv[7];
#pragma unroll
    for (int ct = 0; ct < 7; ++ct) {
      const int col = ct * 16 + n;
      float mval;
      if (col >= 98)      mval = -1e30f;           // kill padded key columns
      else if (row < 98)  mval = mask[(long)mw * 9604 + row * 98 + col];
      else                mval = 0.f;              // padded query rows: finite garbage
      sv[ct] = acc[ct][r] + mval;
    }
    float rmax = sv[0];
#pragma unroll
    for (int ct = 1; ct < 7; ++ct) rmax = fmaxf(rmax, sv[ct]);
#pragma unroll
    for (int off = 8; off > 0; off >>= 1) rmax = fmaxf(rmax, __shfl_xor(rmax, off, 32));
    float rsum = 0.f;
#pragma unroll
    for (int ct = 0; ct < 7; ++ct) { sv[ct] = __expf(sv[ct] - rmax); rsum += sv[ct]; }
#pragma unroll
    for (int off = 8; off > 0; off >>= 1) rsum += __shfl_xor(rsum, off, 32);
    const float inv = 1.f / rsum;
    const int lr = r + 8 * hf;
#pragma unroll
    for (int ct = 0; ct < 7; ++ct) P[wv][lr][ct * 16 + n] = (_Float16)(sv[ct] * inv);
    P[wv][lr][112 + n] = (_Float16)0.f;            // K padding 112..127
  }
  __syncthreads();
  // O = P @ V  (K = 128 padded, N = 32)
  v8f oc0 = {}, oc1 = {};
  const _Float16* vbase = vt + (long)unit * 32 * NTOK;
#pragma unroll
  for (int s = 0; s < 4; ++s) {
    const v16h ap  = load_frag_rm(&P[wv][0][s * 32], 128);
    const v16h bv0 = load_frag_vt(vbase,  0, s * 32);
    const v16h bv1 = load_frag_vt(vbase, 16, s * 32);
    oc0 = WMMA_F16(ap, bv0, oc0);
    oc1 = WMMA_F16(ap, bv1, oc1);
  }
  const int head = unit % 3;
#pragma unroll
  for (int r = 0; r < 8; ++r) {
    const int tok = m0 + r + 8 * hf;
    const long ob = ((long)win * NTOK + tok) * NC + head * 32;
    o[ob + n]      = (_Float16)oc0[r];
    o[ob + 16 + n] = (_Float16)oc1[r];
  }
}

// ------ kernel 4: proj GEMM + window reverse + roll(+shift) + residual ------
__global__ void k_proj(const _Float16* __restrict__ o, const _Float16* __restrict__ wp,
                       const float* __restrict__ bp, const float* __restrict__ x,
                       float* __restrict__ ax) {
  const int lane = threadIdx.x & 31, wv = threadIdx.x >> 5;
  const int n = lane & 15, hf = lane >> 4;
  const int m0 = (blockIdx.x * 4 + wv) * 16;
  prefetch_tile_rows(o + (long)m0 * 96, 96);
  v8f acc[6] = {};
#pragma unroll
  for (int kkk = 0; kkk < 96; kkk += 32) {
    const v16h a = load_frag_rm(o + (long)m0 * 96 + kkk, 96);
#pragma unroll
    for (int t = 0; t < 6; ++t) {
      const v16h b = load_frag_rm(wp + (t * 16) * 96 + kkk, 96);
      acc[t] = WMMA_F16(a, b, acc[t]);
    }
  }
#pragma unroll
  for (int r = 0; r < 8; ++r) {
    const int row = m0 + r + 8 * hf;
    const int win = row / NTOK, tok = row % NTOK;
    if (tok >= 98) continue;
    int b = win >> 10, rem = win & 1023;
    const int wd = rem >> 6; const int wh = (rem >> 3) & 7, ww = rem & 7;
    const int td = tok / 49, r2 = tok % 49, th = r2 / 7, tw = r2 % 7;
    const int d = ((wd << 1) + td + 1) & 31;
    const int h = (wh * 7 + th + 3) % NH;
    const int w = (ww * 7 + tw + 3) % NW;
    const long base = ((((long)b * ND + d) * NH + h) * NW + w) * NC;
#pragma unroll
    for (int t = 0; t < 6; ++t) {
      const int col = t * 16 + n;
      ax[base + col] = x[base + col] + bp[col] + acc[t][r];
    }
  }
}

// ------ kernel 5: LN2 over attn output -> h2 f16 ----------------------------
__global__ void k_ln2(const float* __restrict__ ax, const float* __restrict__ g,
                      const float* __restrict__ bt, _Float16* __restrict__ h2) {
  const int lane = threadIdx.x & 31;
  const int m = blockIdx.x * 4 + (threadIdx.x >> 5);
  if (m >= LTOK) return;
  const float* src = ax + (long)m * NC;
  const float v0 = src[lane], v1 = src[lane + 32], v2 = src[lane + 64];
  float s = v0 + v1 + v2, ss = v0 * v0 + v1 * v1 + v2 * v2;
#pragma unroll
  for (int off = 16; off > 0; off >>= 1) {
    s  += __shfl_xor(s, off, 32);
    ss += __shfl_xor(ss, off, 32);
  }
  const float mean = s * (1.f / 96.f);
  const float rstd = rsqrtf(ss * (1.f / 96.f) - mean * mean + 1e-5f);
  _Float16* dst = h2 + (long)m * NC;
  dst[lane]      = (_Float16)((v0 - mean) * rstd * g[lane]      + bt[lane]);
  dst[lane + 32] = (_Float16)((v1 - mean) * rstd * g[lane + 32] + bt[lane + 32]);
  dst[lane + 64] = (_Float16)((v2 - mean) * rstd * g[lane + 64] + bt[lane + 64]);
}

// ------ kernel 6: FC1 GEMM (M=200704, N=384, K=96) + exact GELU -> hid f16 --
__global__ void k_fc1(const _Float16* __restrict__ h2, const _Float16* __restrict__ w1,
                      const float* __restrict__ b1, _Float16* __restrict__ hid) {
  const int lane = threadIdx.x & 31, wv = threadIdx.x >> 5;
  const int n = lane & 15, hf = lane >> 4;
  const int m0 = (blockIdx.x * 4 + wv) * 16;
  const int n0 = blockIdx.y * 64;
  prefetch_tile_rows(h2 + (long)m0 * 96, 96);
  v8f acc[4] = {};
#pragma unroll
  for (int kkk = 0; kkk < 96; kkk += 32) {
    const v16h a = load_frag_rm(h2 + (long)m0 * 96 + kkk, 96);
#pragma unroll
    for (int t = 0; t < 4; ++t) {
      const v16h b = load_frag_rm(w1 + (n0 + t * 16) * 96 + kkk, 96);
      acc[t] = WMMA_F16(a, b, acc[t]);
    }
  }
#pragma unroll
  for (int t = 0; t < 4; ++t) {
    const int col = n0 + t * 16 + n;
    const float bias = b1[col];
#pragma unroll
    for (int r = 0; r < 8; ++r) {
      float v = acc[t][r] + bias;
      v = 0.5f * v * (1.f + erff(v * 0.70710678118654752f));  // exact GELU
      hid[(long)(m0 + r + 8 * hf) * 384 + col] = (_Float16)v;
    }
  }
}

// ------ kernel 7: FC2 GEMM (M=200704, N=96, K=384) + residual -> out f32 ----
__global__ void k_fc2(const _Float16* __restrict__ hid, const _Float16* __restrict__ w2,
                      const float* __restrict__ b2, const float* __restrict__ ax,
                      float* __restrict__ out) {
  const int lane = threadIdx.x & 31, wv = threadIdx.x >> 5;
  const int n = lane & 15, hf = lane >> 4;
  const int m0 = (blockIdx.x * 4 + wv) * 16;
  v8f acc[6] = {};
#pragma unroll
  for (int kkk = 0; kkk < 384; kkk += 32) {
    if (kkk + 32 < 384)   // prefetch next K-step of the 154 MB hid stream
      prefetch_tile_rows(hid + (long)m0 * 384 + kkk + 32, 384);
    const v16h a = load_frag_rm(hid + (long)m0 * 384 + kkk, 384);
#pragma unroll
    for (int t = 0; t < 6; ++t) {
      const v16h b = load_frag_rm(w2 + (t * 16) * 384 + kkk, 384);
      acc[t] = WMMA_F16(a, b, acc[t]);
    }
  }
#pragma unroll
  for (int t = 0; t < 6; ++t) {
    const int col = t * 16 + n;
    const float bias = b2[col];
#pragma unroll
    for (int r = 0; r < 8; ++r) {
      const long idx = (long)(m0 + r + 8 * hf) * NC + col;
      out[idx] = ax[idx] + bias + acc[t][r];
    }
  }
}

// --------------------------- launcher ---------------------------------------
extern "C" void kernel_launch(void* const* d_in, const int* in_sizes, int n_in,
                              void* d_out, int out_size, void* d_ws, size_t ws_size,
                              hipStream_t stream) {
  (void)in_sizes; (void)n_in; (void)out_size; (void)ws_size;
  const float* x     = (const float*)d_in[0];
  const float* mask  = (const float*)d_in[1];
  const float* n1g   = (const float*)d_in[2];
  const float* n1b   = (const float*)d_in[3];
  const float* qkvw  = (const float*)d_in[4];
  const float* qkvb  = (const float*)d_in[5];
  const float* projw = (const float*)d_in[6];
  const float* projb = (const float*)d_in[7];
  const float* n2g   = (const float*)d_in[8];
  const float* n2b   = (const float*)d_in[9];
  const float* fc1w  = (const float*)d_in[10];
  const float* fc1b  = (const float*)d_in[11];
  const float* fc2w  = (const float*)d_in[12];
  const float* fc2b  = (const float*)d_in[13];
  float* out = (float*)d_out;

  // workspace layout (phase-aliased)
  char* ws = (char*)d_ws;
  const size_t SZ_TOKF16 = (size_t)MROWS * NC * sizeof(_Float16);  // 44,040,192
  _Float16* xw  = (_Float16*)(ws + 0);                       // phase 1-2
  _Float16* q   = (_Float16*)(ws + 1 * SZ_TOKF16);           // phase 2-3
  _Float16* kk  = (_Float16*)(ws + 2 * SZ_TOKF16);
  _Float16* vt  = (_Float16*)(ws + 3 * SZ_TOKF16);
  _Float16* o   = (_Float16*)(ws + 4 * SZ_TOKF16);           // phase 3-4
  float*    ax  = (float*)   (ws + 5 * SZ_TOKF16);           // phase 4-end (77 MB)
  _Float16* h2  = (_Float16*)(ws + 4 * SZ_TOKF16);           // aliases o (dead)
  _Float16* hid = (_Float16*)(ws + 0);                       // aliases xw/q/k (dead)
  char* wbase   = ws + 5 * SZ_TOKF16 + (size_t)LTOK * NC * sizeof(float);
  _Float16* wq  = (_Float16*)wbase;
  _Float16* wp  = wq + 27648;
  _Float16* w1  = wp + 9216;
  _Float16* w2  = w1 + 36864;

  k_cvt_w    <<<144, 256, 0, stream>>>(qkvw, projw, fc1w, fc2w, wq, wp, w1, w2);
  k_ln1_window<<<MROWS / 4, 128, 0, stream>>>(x, n1g, n1b, xw);
  k_qkv      <<<dim3(MROWS / 64, 6), 128, 0, stream>>>(xw, wq, qkvb, q, kk, vt);
  k_attn     <<<NWIN * 3, 224, 0, stream>>>(q, kk, vt, mask, o);
  k_proj     <<<MROWS / 64, 128, 0, stream>>>(o, wp, projb, x, ax);
  k_ln2      <<<LTOK / 4, 128, 0, stream>>>(ax, n2g, n2b, h2);
  k_fc1      <<<dim3(LTOK / 64, 6), 128, 0, stream>>>(h2, w1, fc1b, hid);
  k_fc2      <<<LTOK / 64, 128, 0, stream>>>(hid, w2, fc2b, ax, out);
}
